// nmr_mpnn_BASELINE_59536836657944
// MI455X (gfx1250) — compile-verified, matrix operand order checked by reading, not run on hardware
//
#include <hip/hip_runtime.h>
#include <hip/hip_bf16.h>

typedef __attribute__((ext_vector_type(16))) __bf16       v16bf;
typedef __attribute__((ext_vector_type(8)))  float        v8f;
typedef __attribute__((ext_vector_type(4)))  unsigned int u32x4;
typedef __attribute__((ext_vector_type(8)))  int          i32x8;
typedef __attribute__((ext_vector_type(4)))  int          i32x4;

#define NNODE  4096
#define NEDGE  16384
#define NIDIM  64
#define EINDIM 16
#define FDIM   128
#define HDIM   512
#define MROWS  2048
#define NSTEPS 5
#define KEDGE  2176   /* F*EI + F : kron part + bias-matrix part */
#define NKCH   (KEDGE / 32)   /* 68 K-chunks of 32 */
#define G3     384

// CDNA5 16x32 bf16 fragment: lane holds M/N = lane&15; K-halves are
// [base..base+7] and [base+16..base+23], base = (lane&16)?8:0.
// => a fragment is two aligned 16B loads from a K-contiguous row.
union Frag { v16bf bf; u32x4 q[2]; };

__device__ __forceinline__ __bf16 f2bf(float x) { return (__bf16)x; }

__device__ __forceinline__ v16bf load_frag(const __bf16* rowp, int kbase) {
  Frag f;
  f.q[0] = *(const u32x4*)(rowp + kbase);
  f.q[1] = *(const u32x4*)(rowp + kbase + 16);
  return f.bf;
}

// ---------------------------------------------------------------------------
// TDM: DMA one 32(K) x 128(row) bf16 tile of Wpc [128][KEDGE] into LDS.
// ---------------------------------------------------------------------------
__device__ __forceinline__ void tdm_load_chunk(const __bf16* gaddr,
                                               unsigned lds_byte_addr) {
  unsigned long long ga = (unsigned long long)(size_t)gaddr;
  u32x4 g0;
  g0[0] = 1u;                                           // count=1, user mode
  g0[1] = lds_byte_addr;                                // D#.lds_addr
  g0[2] = (unsigned)(ga & 0xFFFFFFFFu);                 // global_addr[31:0]
  g0[3] = (unsigned)((ga >> 32) & 0x1FFFFFFu) | (2u << 30); // addr[56:32]|type=2
  i32x8 g1;
  g1[0] = (int)(1u << 16);          // data_size=1 (2 bytes), no multicast
  g1[1] = (int)(KEDGE << 16);       // tensor_dim0[15:0] @ bits 63:48
  g1[2] = (int)(128u << 16);        // tensor_dim1[15:0] @ bits 95:80 (dim0 hi=0)
  g1[3] = (int)(32u << 16);         // tile_dim0=32 @ bits 127:112
  g1[4] = (int)128;                 // tile_dim1=128, tile_dim2=0
  g1[5] = (int)KEDGE;               // tensor_dim0_stride low32
  g1[6] = 0;
  g1[7] = 0;
  i32x4 z4 = {0, 0, 0, 0};
#if defined(__clang_major__) && (__clang_major__ >= 23)
  i32x8 z8 = {0, 0, 0, 0, 0, 0, 0, 0};
  __builtin_amdgcn_tensor_load_to_lds(g0, g1, z4, z4, z8, 0);
#else
  __builtin_amdgcn_tensor_load_to_lds(g0, g1, z4, z4, 0);
#endif
}

// ---------------------------------------------------------------------------
// Weight packing (bf16).  All K-major per output row:
//   Wpc   [o][kk]  kk<2048: W_edge[(f*128+o)*16+k], kk=f*16+k ; else b_edge
//   Wproj [o][k], Wih/Whh [g][k], W1 [o][k], W2 [o][k]  (natural layouts)
// ---------------------------------------------------------------------------
__global__ __launch_bounds__(256) void k_prep(
    const float* __restrict__ W_proj, const float* __restrict__ W_edge,
    const float* __restrict__ b_edge, const float* __restrict__ W_ih,
    const float* __restrict__ W_hh,  const float* __restrict__ W1,
    const float* __restrict__ W2,
    __bf16* __restrict__ Wpc, __bf16* __restrict__ Wproj_bf,
    __bf16* __restrict__ Wih_bf, __bf16* __restrict__ Whh_bf,
    __bf16* __restrict__ W1_bf, __bf16* __restrict__ W2_bf)
{
  const int nWp = FDIM * KEDGE;
  const int nPr = FDIM * NIDIM;
  const int nIh = G3 * FDIM;
  const int n1  = HDIM * (NIDIM + FDIM);
  const int n2  = HDIM * HDIM;
  const int total = nWp + nPr + nIh + nIh + n1 + n2;
  for (int i = blockIdx.x * blockDim.x + threadIdx.x; i < total;
       i += gridDim.x * blockDim.x) {
    int t = i;
    if (t < nWp) {
      int o = t / KEDGE, kk = t % KEDGE;
      float v = (kk < FDIM * EINDIM)
                    ? W_edge[(((kk >> 4) * FDIM) + o) * EINDIM + (kk & 15)]
                    : b_edge[(kk - FDIM * EINDIM) * FDIM + o];
      Wpc[t] = f2bf(v);
      continue;
    }
    t -= nWp;
    if (t < nPr) { Wproj_bf[t] = f2bf(W_proj[t]); continue; }
    t -= nPr;
    if (t < nIh) { Wih_bf[t] = f2bf(W_ih[t]); continue; }
    t -= nIh;
    if (t < nIh) { Whh_bf[t] = f2bf(W_hh[t]); continue; }
    t -= nIh;
    if (t < n1)  { W1_bf[t] = f2bf(W1[t]); continue; }
    t -= n1;
    W2_bf[t] = f2bf(W2[t]);
  }
}

// ---------------------------------------------------------------------------
// h0 = relu(node_attr @ W_proj^T + b_proj)
// ---------------------------------------------------------------------------
__global__ __launch_bounds__(256) void k_h0(
    const float* __restrict__ node_attr, const __bf16* __restrict__ Wproj_bf,
    const float* __restrict__ b_proj, float* __restrict__ h0)
{
  __shared__ alignas(16) __bf16 sa[16][NIDIM];
  const int tid = threadIdx.x, lane = tid & 31, wave = tid >> 5;
  const int rbase = blockIdx.x * 16;
  for (int i = tid; i < 16 * NIDIM; i += 256)
    sa[i >> 6][i & 63] = f2bf(node_attr[(rbase + (i >> 6)) * NIDIM + (i & 63)]);
  __syncthreads();
  const int n = lane & 15, m = lane & 15, base = (lane & 16) ? 8 : 0;
  const int col = wave * 16 + n;
  const __bf16* Bw = Wproj_bf + col * NIDIM;
  v8f acc = {};
#pragma unroll
  for (int k0 = 0; k0 < NIDIM; k0 += 32) {
    v16bf a = load_frag(&sa[m][0], k0 + base);
    v16bf b = load_frag(Bw, k0 + base);
    acc = __builtin_amdgcn_wmma_f32_16x16x32_bf16(false, a, false, b,
                                                  (short)0, acc, false, false);
  }
#pragma unroll
  for (int v = 0; v < 8; ++v)
    h0[(rbase + v + base) * FDIM + col] = fmaxf(acc[v] + b_proj[col], 0.f);
}

// ---------------------------------------------------------------------------
// Fused NNConv message GEMM + scatter.  Block = 128 edges (8 waves x 16-edge
// row tiles); each wave sweeps all 8 column tiles (8 accumulators) so one
// A-fragment feeds 8 WMMAs.  B chunks (32K x 128 cols) stream in via the
// Tensor Data Mover, double-buffered, tracked with TENSORcnt.
// ---------------------------------------------------------------------------
__global__ __launch_bounds__(256) void k_edge_msg(
    const float* __restrict__ h, const float* __restrict__ edge_attr,
    const int* __restrict__ src, const int* __restrict__ dst,
    const __bf16* __restrict__ Wpc, float* __restrict__ agg)
{
  __shared__ alignas(16) __bf16 sh_h[128][FDIM];      // 32 KB gathered h_src
  __shared__ alignas(16) float  sh_e[128][EINDIM];    //  8 KB edge_attr
  __shared__ alignas(16) __bf16 sh_B[2][FDIM][32];    // 16 KB TDM dst [col][krel]
  const int tid = threadIdx.x, lane = tid & 31, wave = tid >> 5;
  const int ebase = blockIdx.x * 128;

  if (wave == 0)   // kick off chunk 0 while we stage the gathers
    tdm_load_chunk(Wpc, (unsigned)(size_t)&sh_B[0][0][0]);

  for (int i = tid; i < 128 * FDIM; i += 256) {
    int r = i >> 7, f = i & 127;
    sh_h[r][f] = f2bf(h[src[ebase + r] * FDIM + f]);
  }
  for (int i = tid; i < 128 * EINDIM; i += 256)
    sh_e[i >> 4][i & 15] = edge_attr[(ebase + (i >> 4)) * EINDIM + (i & 15)];

  const int n = lane & 15, base = (lane & 16) ? 8 : 0;
  const int r = wave * 16 + (lane & 15);              // A row owned by lane

  v8f acc[8] = {};
  float er[EINDIM];
  for (int it = 0; it < NKCH; ++it) {
    const int k0 = it << 5;
    if (wave == 0) __builtin_amdgcn_s_wait_tensorcnt(0);
    __syncthreads();                                  // buffer it ready for all
    if (it == 0) {                                    // edge_attr row now staged
#pragma unroll
      for (int i = 0; i < EINDIM; ++i) er[i] = sh_e[r][i];
    }
    if (wave == 0 && it + 1 < NKCH)
      tdm_load_chunk(Wpc + (it + 1) * 32,
                     (unsigned)(size_t)&sh_B[(it + 1) & 1][0][0]);
    // --- build A fragment for this 32-K chunk ---
    Frag af;
    if (k0 < FDIM * EINDIM) {                         // Kronecker part
      const int f0 = k0 >> 4;
      const float h0v = (float)sh_h[r][f0];
      const float h1v = (float)sh_h[r][f0 + 1];
#pragma unroll
      for (int j = 0; j < 16; ++j)
        af.bf[j] = f2bf(((j < 8) ? h0v : h1v) * er[base + (j & 7)]);
    } else {                                          // bias-matrix part: h itself
      af.q[0] = *(const u32x4*)&sh_h[r][(k0 - FDIM * EINDIM) + base];
      af.q[1] = *(const u32x4*)&sh_h[r][(k0 - FDIM * EINDIM) + base + 16];
    }
    const __bf16* bbuf = &sh_B[it & 1][0][0];
#pragma unroll
    for (int ct = 0; ct < 8; ++ct) {
      v16bf b = load_frag(bbuf + (ct * 16 + n) * 32, base);
      acc[ct] = __builtin_amdgcn_wmma_f32_16x16x32_bf16(
          false, af.bf, false, b, (short)0, acc[ct], false, false);
    }
  }
  // --- scatter-add into agg[dst] ---
  int drow[8];
#pragma unroll
  for (int v = 0; v < 8; ++v)
    drow[v] = dst[ebase + wave * 16 + v + base] * FDIM;
#pragma unroll
  for (int ct = 0; ct < 8; ++ct)
#pragma unroll
    for (int v = 0; v < 8; ++v)
      atomicAdd(&agg[drow[v] + ct * 16 + n], acc[ct][v]);
}

// ---------------------------------------------------------------------------
// GRU step: x = relu(agg + b_conv); gx = x@Wih^T; gh = h@Whh^T; gate math.
// ---------------------------------------------------------------------------
__global__ __launch_bounds__(256) void k_gru(
    const float* __restrict__ agg, const float* __restrict__ h,
    const __bf16* __restrict__ Wih_bf, const __bf16* __restrict__ Whh_bf,
    const float* __restrict__ b_conv, const float* __restrict__ b_ih,
    const float* __restrict__ b_hh, float* __restrict__ h_out)
{
  __shared__ alignas(16) __bf16 sx[16][FDIM];
  __shared__ alignas(16) __bf16 shh[16][FDIM];
  __shared__ float gx[16][G3];
  __shared__ float gh[16][G3];
  const int tid = threadIdx.x, lane = tid & 31, wave = tid >> 5;
  const int rbase = blockIdx.x * 16;

  for (int i = tid; i < 16 * FDIM; i += 256) {
    int rr = i >> 7, f = i & 127;
    float xv = agg[(rbase + rr) * FDIM + f] + b_conv[f];
    sx[rr][f]  = f2bf(fmaxf(xv, 0.f));
    shh[rr][f] = f2bf(h[(rbase + rr) * FDIM + f]);
  }
  __syncthreads();

  const int n = lane & 15, m = lane & 15, base = (lane & 16) ? 8 : 0;
  for (int t = wave; t < 48; t += 8) {                // wave-uniform (6 iters)
    const bool isX = t < 24;
    const int ct = isX ? t : t - 24;
    const __bf16* A0 = isX ? &sx[m][0] : &shh[m][0];
    const __bf16* Bw = (isX ? Wih_bf : Whh_bf) + (ct * 16 + n) * FDIM;
    v8f acc = {};
#pragma unroll
    for (int k0 = 0; k0 < FDIM; k0 += 32) {
      v16bf a = load_frag(A0, k0 + base);
      v16bf b = load_frag(Bw, k0 + base);
      acc = __builtin_amdgcn_wmma_f32_16x16x32_bf16(false, a, false, b,
                                                    (short)0, acc, false, false);
    }
    float* G = isX ? &gx[0][0] : &gh[0][0];
#pragma unroll
    for (int v = 0; v < 8; ++v)
      G[(v + base) * G3 + ct * 16 + n] = acc[v];
  }
  __syncthreads();

  for (int i = tid; i < 16 * FDIM; i += 256) {
    int rr = i >> 7, f = i & 127;
    float xr = gx[rr][f]            + b_ih[f];
    float xz = gx[rr][FDIM + f]     + b_ih[FDIM + f];
    float xn = gx[rr][2 * FDIM + f] + b_ih[2 * FDIM + f];
    float hr = gh[rr][f]            + b_hh[f];
    float hz = gh[rr][FDIM + f]     + b_hh[FDIM + f];
    float hn = gh[rr][2 * FDIM + f] + b_hh[2 * FDIM + f];
    float rg   = 1.f / (1.f + __expf(-(xr + hr)));
    float zg   = 1.f / (1.f + __expf(-(xz + hz)));
    float cand = tanhf(xn + rg * hn);
    float hv   = h[(rbase + rr) * FDIM + f];
    h_out[(rbase + rr) * FDIM + f] = (1.f - zg) * cand + zg * hv;
  }
}

// ---------------------------------------------------------------------------
// Readout layer 1: gather masks, concat [node_attr | h] (K=192) -> relu GEMM
// ---------------------------------------------------------------------------
__global__ __launch_bounds__(256) void k_read1(
    const int* __restrict__ masks, const float* __restrict__ node_attr,
    const float* __restrict__ h, const __bf16* __restrict__ W1_bf,
    const float* __restrict__ b1, __bf16* __restrict__ y1)
{
  __shared__ alignas(16) __bf16 sa[16][NIDIM + FDIM];
  const int tid = threadIdx.x, lane = tid & 31, wave = tid >> 5;
  const int rbase = blockIdx.x * 16;
  for (int i = tid; i < 16 * (NIDIM + FDIM); i += 256) {
    int rr = i / (NIDIM + FDIM), c = i % (NIDIM + FDIM);
    int node = masks[rbase + rr];
    float v = (c < NIDIM) ? node_attr[node * NIDIM + c]
                          : h[node * FDIM + (c - NIDIM)];
    sa[rr][c] = f2bf(v);
  }
  __syncthreads();
  const int n = lane & 15, m = lane & 15, base = (lane & 16) ? 8 : 0;
  for (int t = wave; t < HDIM / 16; t += 8) {
    const int col = t * 16 + n;
    const __bf16* Bw = W1_bf + col * (NIDIM + FDIM);
    v8f acc = {};
#pragma unroll
    for (int k0 = 0; k0 < NIDIM + FDIM; k0 += 32) {
      v16bf a = load_frag(&sa[m][0], k0 + base);
      v16bf b = load_frag(Bw, k0 + base);
      acc = __builtin_amdgcn_wmma_f32_16x16x32_bf16(false, a, false, b,
                                                    (short)0, acc, false, false);
    }
#pragma unroll
    for (int v = 0; v < 8; ++v)
      y1[(rbase + v + base) * HDIM + col] = f2bf(fmaxf(acc[v] + b1[col], 0.f));
  }
}

// ---------------------------------------------------------------------------
// Readout layer 2: (2048x512)x(512x512) relu GEMM
// ---------------------------------------------------------------------------
__global__ __launch_bounds__(256) void k_read2(
    const __bf16* __restrict__ y1, const __bf16* __restrict__ W2_bf,
    const float* __restrict__ b2, __bf16* __restrict__ y2)
{
  __shared__ alignas(16) __bf16 sa[16][HDIM];
  const int tid = threadIdx.x, lane = tid & 31, wave = tid >> 5;
  const int rbase = blockIdx.x * 16;
  for (int i = tid; i < 16 * HDIM; i += 256)
    sa[i >> 9][i & 511] = y1[(rbase + (i >> 9)) * HDIM + (i & 511)];
  __syncthreads();
  const int n = lane & 15, m = lane & 15, base = (lane & 16) ? 8 : 0;
  for (int t = wave; t < HDIM / 16; t += 8) {
    const int col = t * 16 + n;
    const __bf16* Bw = W2_bf + col * HDIM;
    v8f acc = {};
#pragma unroll 4
    for (int k0 = 0; k0 < HDIM; k0 += 32) {
      v16bf a = load_frag(&sa[m][0], k0 + base);
      v16bf b = load_frag(Bw, k0 + base);
      acc = __builtin_amdgcn_wmma_f32_16x16x32_bf16(false, a, false, b,
                                                    (short)0, acc, false, false);
    }
#pragma unroll
    for (int v = 0; v < 8; ++v)
      y2[(rbase + v + base) * HDIM + col] = f2bf(fmaxf(acc[v] + b2[col], 0.f));
  }
}

// ---------------------------------------------------------------------------
// Readout layer 3: GEMV, one wave per row
// ---------------------------------------------------------------------------
__global__ __launch_bounds__(256) void k_read3(
    const __bf16* __restrict__ y2, const float* __restrict__ W3,
    const float* __restrict__ b3, float* __restrict__ out)
{
  const int wv = (blockIdx.x * 256 + threadIdx.x) >> 5;
  const int lane = threadIdx.x & 31;
  if (wv >= MROWS) return;
  float s = 0.f;
  for (int k = lane; k < HDIM; k += 32)
    s += (float)y2[wv * HDIM + k] * W3[k];
#pragma unroll
  for (int off = 16; off > 0; off >>= 1) s += __shfl_xor(s, off, 32);
  if (lane == 0) out[wv] = s + b3[0];
}

// ---------------------------------------------------------------------------
extern "C" void kernel_launch(void* const* d_in, const int* in_sizes, int n_in,
                              void* d_out, int out_size, void* d_ws,
                              size_t ws_size, hipStream_t stream)
{
  (void)in_sizes; (void)n_in; (void)out_size; (void)ws_size;
  const float* node_attr = (const float*)d_in[0];
  const float* edge_attr = (const float*)d_in[1];
  const int*   src   = (const int*)d_in[2];
  const int*   dst   = (const int*)d_in[3];
  const int*   masks = (const int*)d_in[4];
  const float* W_proj = (const float*)d_in[5];
  const float* b_proj = (const float*)d_in[6];
  const float* W_edge = (const float*)d_in[7];
  const float* b_edge = (const float*)d_in[8];
  const float* b_conv = (const float*)d_in[9];
  const float* W_ih = (const float*)d_in[10];
  const float* W_hh = (const float*)d_in[11];
  const float* b_ih = (const float*)d_in[12];
  const float* b_hh = (const float*)d_in[13];
  const float* W1 = (const float*)d_in[14];
  const float* b1 = (const float*)d_in[15];
  const float* W2 = (const float*)d_in[16];
  const float* b2 = (const float*)d_in[17];
  const float* W3 = (const float*)d_in[18];
  const float* b3 = (const float*)d_in[19];

  char* p = (char*)d_ws;
  auto alloc = [&](size_t bytes) -> char* {
    char* r = p; p += (bytes + 255) & ~(size_t)255; return r;
  };
  float*  hA  = (float*)alloc((size_t)NNODE * FDIM * 4);
  float*  hB  = (float*)alloc((size_t)NNODE * FDIM * 4);
  float*  agg = (float*)alloc((size_t)NNODE * FDIM * 4);
  __bf16* Wpc      = (__bf16*)alloc((size_t)FDIM * KEDGE * 2);
  __bf16* Wproj_bf = (__bf16*)alloc((size_t)FDIM * NIDIM * 2);
  __bf16* Wih_bf   = (__bf16*)alloc((size_t)G3 * FDIM * 2);
  __bf16* Whh_bf   = (__bf16*)alloc((size_t)G3 * FDIM * 2);
  __bf16* W1_bf    = (__bf16*)alloc((size_t)HDIM * (NIDIM + FDIM) * 2);
  __bf16* W2_bf    = (__bf16*)alloc((size_t)HDIM * HDIM * 2);
  __bf16* y1       = (__bf16*)alloc((size_t)MROWS * HDIM * 2);
  __bf16* y2       = (__bf16*)alloc((size_t)MROWS * HDIM * 2);

  k_prep<<<1024, 256, 0, stream>>>(W_proj, W_edge, b_edge, W_ih, W_hh, W1, W2,
                                   Wpc, Wproj_bf, Wih_bf, Whh_bf, W1_bf, W2_bf);
  k_h0<<<NNODE / 16, 256, 0, stream>>>(node_attr, Wproj_bf, b_proj, hA);

  float* hc = hA;
  float* hn = hB;
  for (int s = 0; s < NSTEPS; ++s) {
    (void)hipMemsetAsync(agg, 0, (size_t)NNODE * FDIM * 4, stream);
    k_edge_msg<<<NEDGE / 128, 256, 0, stream>>>(hc, edge_attr, src, dst, Wpc, agg);
    k_gru<<<NNODE / 16, 256, 0, stream>>>(agg, hc, Wih_bf, Whh_bf, b_conv,
                                          b_ih, b_hh, hn);
    float* t = hc; hc = hn; hn = t;
  }

  k_read1<<<MROWS / 16, 256, 0, stream>>>(masks, node_attr, hc, W1_bf, b1, y1);
  k_read2<<<MROWS / 16, 256, 0, stream>>>(y1, W2_bf, b2, y2);
  k_read3<<<MROWS / 8, 256, 0, stream>>>(y2, W3, b3, (float*)d_out);
}
